// TriangleAttentionStartingNode_31645319037643
// MI455X (gfx1250) — matrix operand order repaired and checked by reading, not compiled, not run on hardware
//
#include <hip/hip_runtime.h>
#include <hip/hip_bf16.h>

// ---------------------------------------------------------------------------
// TriangleAttentionStartingNode for MI455X (gfx1250, wave32, WMMA 16x16x32 f16)
//
// k_prep : weights f32 -> f16, pair-packed for 1-dword B-fragment loads
// k_proj : LayerNorm(f32) + tri-bias(f32) + QKG projection (WMMA) + V^T
// k_attn : flash attention per (row, head); logits live in registers only
// k_out  : output projection (WMMA) + bias + residual
// ---------------------------------------------------------------------------
#define NPTS  65536   // B*N*N points
#define NSEQ  256
#define DP    128
#define NHEAD 4
#define CDIM  32
#define QKG   384     // qkvg row stride: q|k|g

typedef __attribute__((ext_vector_type(16))) _Float16 v16h;
typedef __attribute__((ext_vector_type(8)))  float    v8f;

union Frag16 { v16h v; unsigned u[8]; };

__device__ __forceinline__ v8f zero8() {
  v8f z = {0.f,0.f,0.f,0.f,0.f,0.f,0.f,0.f};
  return z;
}

__device__ __forceinline__ v8f wmma_f16(v16h a, v16h b, v8f c) {
  return __builtin_amdgcn_wmma_f32_16x16x32_f16(false, a, false, b, (short)0, c, false, false);
}

// A-fragment 16x32 f16: lane m = lane&15, VGPR r holds K pair
//   k(r,hi) = 2r + 16*hi   (contiguous order, matches B fragments)
// rowptr must already include the per-lane row offset (m * stride).
__device__ __forceinline__ v16h load_a16(const _Float16* rowptr, int hi) {
  Frag16 f;
#pragma unroll
  for (int r = 0; r < 8; ++r) {
    int k = 2*r + 16*hi;
    f.u[r] = *(const unsigned*)(rowptr + k);
  }
  return f.v;
}

// B-fragment 32x16 f16: lane n = lane&15, K pair k(r,hi) = 2r + 16*hi
// (this code's contiguous order: lanes 0-15 hold K=0..15, lanes 16-31 hold K=16..31)
// Source where (k,k+1) are contiguous for fixed n: one dword load per VGPR.
__device__ __forceinline__ v16h load_b_ct(const _Float16* colptr, int hi) {
  Frag16 f;
#pragma unroll
  for (int r = 0; r < 8; ++r)
    f.u[r] = *(const unsigned*)(colptr + 2*r + 16*hi);
  return f.v;
}

// Pair-packed weights W_pk[p][n][2] with p=k/2 (built by k_prep).
__device__ __forceinline__ v16h load_b_pk(const _Float16* wpk, int ncols2, int col, int hi) {
  Frag16 f;
#pragma unroll
  for (int r = 0; r < 8; ++r)
    f.u[r] = *(const unsigned*)(wpk + (size_t)(r + 8*hi)*ncols2 + col*2);
  return f.v;
}

// ---------------------------------------------------------------------------
// Kernel 0: one-time weight convert+pack to f16 (L2-resident afterwards).
// ---------------------------------------------------------------------------
__global__ void k_prep(const float* __restrict__ wq, const float* __restrict__ wk,
                       const float* __restrict__ wv, const float* __restrict__ wg,
                       const float* __restrict__ wo,
                       _Float16* __restrict__ Wpk, _Float16* __restrict__ Wopk) {
  int t = blockIdx.x * 256 + threadIdx.x;
  if (t < 128*512) {
    int k = t >> 9, n = t & 511;
    int g = n >> 7, nl = n & 127;
    const float* w = (g == 0) ? wq : (g == 1) ? wk : (g == 2) ? wv : wg;
    Wpk[(k >> 1)*1024 + n*2 + (k & 1)] = (_Float16)w[k*128 + nl];
  } else {
    int t2 = t - 128*512;
    if (t2 < 128*128) {
      int k = t2 >> 7, n = t2 & 127;
      Wopk[(k >> 1)*256 + n*2 + (k & 1)] = (_Float16)wo[k*128 + n];
    }
  }
}

// ---------------------------------------------------------------------------
// Kernel 1: LayerNorm + tri-bias + QKVG projection.
// 256 threads, 32 points/block. Wave w: group = w>>1 (q,k,v,g), Mtile = w&1.
// q (scaled), k, g (sigmoided) -> qkvg[point][384]; v -> vt[i][h][c][j].
// ---------------------------------------------------------------------------
__global__ __launch_bounds__(256) void k_proj(
    const float* __restrict__ Zraw, const float* __restrict__ lnw,
    const float* __restrict__ lnb, const float* __restrict__ wb,
    const float* __restrict__ bg, const _Float16* __restrict__ Wpk,
    _Float16* __restrict__ qkvg, _Float16* __restrict__ vt,
    float* __restrict__ trib) {
  __shared__ __align__(16) _Float16 Zln[32][DP];
  const int tid = threadIdx.x;
  const int pt0 = blockIdx.x * 32;

  // ---- LayerNorm: 8 lanes per point, 16 dims per lane ----
  {
    int pl = tid >> 3, seg = tid & 7, d0 = seg * 16;
    int pt = pt0 + pl;
    const float* x = Zraw + (size_t)pt * DP + d0;
    float xv[16];
#pragma unroll
    for (int e = 0; e < 16; ++e) xv[e] = x[e];
    float s = 0.f, s2 = 0.f;
#pragma unroll
    for (int e = 0; e < 16; ++e) { s += xv[e]; s2 += xv[e]*xv[e]; }
#pragma unroll
    for (int m = 1; m <= 4; m <<= 1) { s += __shfl_xor(s, m, 32); s2 += __shfl_xor(s2, m, 32); }
    float mean = s * (1.0f/128.0f);
    float var  = s2 * (1.0f/128.0f) - mean*mean;
    float rstd = rsqrtf(var + 1e-5f);
    float tb[NHEAD] = {0.f, 0.f, 0.f, 0.f};
#pragma unroll
    for (int e = 0; e < 16; ++e) {
      int d = d0 + e;
      float z = (xv[e] - mean) * rstd * lnw[d] + lnb[d];
      Zln[pl][d] = (_Float16)z;
#pragma unroll
      for (int h = 0; h < NHEAD; ++h) tb[h] += z * wb[d*NHEAD + h];
    }
#pragma unroll
    for (int m = 1; m <= 4; m <<= 1)
#pragma unroll
      for (int h = 0; h < NHEAD; ++h) tb[h] += __shfl_xor(tb[h], m, 32);
    if (seg == 0) {
#pragma unroll
      for (int h = 0; h < NHEAD; ++h) trib[(size_t)h*NPTS + pt] = tb[h];
    }
  }
  __syncthreads();

  // ---- Projection GEMM: [16 x 128] x [128 x 128] per wave ----
  const int wv   = tid >> 5;
  const int lane = tid & 31, lo = lane & 15, hi = lane >> 4;
  const int gidx = wv >> 1;        // 0=q 1=k 2=v 3=g
  const int mt   = wv & 1;
  v8f acc[8];
#pragma unroll
  for (int nt = 0; nt < 8; ++nt) acc[nt] = zero8();
  const _Float16* arow = &Zln[mt*16 + lo][0];
#pragma unroll
  for (int ks = 0; ks < 4; ++ks) {
    v16h a = load_a16(arow + ks*32, hi);
#pragma unroll
    for (int nt = 0; nt < 8; ++nt) {
      v16h b = load_b_pk(Wpk + (size_t)(ks*16)*1024, 1024, gidx*128 + nt*16 + lo, hi);
      acc[nt] = wmma_f16(a, b, acc[nt]);
    }
  }
  const float qscale = 0.17677669529663687f;   // 32^-0.5
  const int prow = pt0 + mt*16;
#pragma unroll
  for (int nt = 0; nt < 8; ++nt) {
    int col = nt*16 + lo;                      // column within group (= h*32+c)
    float bgv = (gidx == 3) ? bg[col] : 0.0f;
#pragma unroll
    for (int r = 0; r < 8; ++r) {
      float vv = acc[nt][r];
      int pt = prow + r + 8*hi;
      if (gidx == 2) {
        // V transposed: vt[((i*4 + h)*32 + c)*256 + j]
        int i = pt >> 8, j = pt & 255;
        vt[((size_t)(i*NHEAD + (col >> 5))*CDIM + (col & 31))*NSEQ + j] = (_Float16)vv;
      } else {
        if (gidx == 0) vv *= qscale;
        if (gidx == 3) vv = 1.0f / (1.0f + __expf(-(vv + bgv)));
        int ofs = (gidx == 3) ? 256 : gidx*128;
        qkvg[(size_t)pt*QKG + ofs + col] = (_Float16)vv;
      }
    }
  }
}

// ---------------------------------------------------------------------------
// Kernel 2: flash attention per (row i, head h). 512 threads = 16 waves,
// each wave owns one 16-row query tile; logits never touch HBM.
// ---------------------------------------------------------------------------
__global__ __launch_bounds__(512) void k_attn(
    const _Float16* __restrict__ qkvg, const _Float16* __restrict__ vt,
    const float* __restrict__ trib, const float* __restrict__ zmask,
    _Float16* __restrict__ og) {
  __shared__ __align__(16) _Float16 Pbuf[16][512];   // per-wave 16x32 P tile
  const int i = blockIdx.x, h = blockIdx.y;
  const int tid = threadIdx.x;
  const int wv = tid >> 5, lane = tid & 31, lo = lane & 15, hi = lane >> 4;
  const int q0 = wv * 16;
  const _Float16* qb  = qkvg + (size_t)i * NSEQ * QKG;
  const _Float16* vtb = vt + (size_t)(i*NHEAD + h) * CDIM * NSEQ;

  v16h aq = load_a16(qb + (size_t)(q0 + lo)*QKG + h*CDIM, hi);   // Q fragment
  v8f o0 = zero8(), o1 = zero8();
  float mr[8], lr[8];
#pragma unroll
  for (int r = 0; r < 8; ++r) { mr[r] = -3.0e38f; lr[r] = 0.f; }

  const float* tbase = trib + (size_t)h * NPTS;
  for (int kc = 0; kc < 8; ++kc) {
    int kb = kc * 32;
    // prefetch next chunk's K rows and V cols while this chunk computes
    if (kc < 7) {
      __builtin_prefetch(qb + (size_t)(kb + 32 + lane)*QKG + 128 + h*CDIM, 0, 1);
      __builtin_prefetch(vtb + (size_t)lane*NSEQ + kb + 32, 0, 1);
    }
    // S = Q K^T   (two 16x16 N-tiles, one wmma each since Kdim=32)
    v16h bk0 = load_b_ct(qb + (size_t)(kb + lo)*QKG      + 128 + h*CDIM, hi);
    v16h bk1 = load_b_ct(qb + (size_t)(kb + 16 + lo)*QKG + 128 + h*CDIM, hi);
    v8f s0 = wmma_f16(aq, bk0, zero8());
    v8f s1 = wmma_f16(aq, bk1, zero8());
    // bias: mask over keys + triangle bias (both f32 from HBM/L2)
    float mb0 = 1e9f * (zmask[i*NSEQ + kb + lo] - 1.0f);
    float mb1 = 1e9f * (zmask[i*NSEQ + kb + 16 + lo] - 1.0f);
#pragma unroll
    for (int r = 0; r < 8; ++r) {
      int qr = q0 + r + 8*hi;
      s0[r] += tbase[(size_t)qr*NSEQ + kb + lo] + mb0;
      s1[r] += tbase[(size_t)qr*NSEQ + kb + 16 + lo] + mb1;
    }
    // online softmax (row = r + 8*hi lives in one 16-lane half -> xor<=8)
#pragma unroll
    for (int r = 0; r < 8; ++r) {
      float mx = fmaxf(s0[r], s1[r]);
#pragma unroll
      for (int m = 1; m <= 8; m <<= 1) mx = fmaxf(mx, __shfl_xor(mx, m, 32));
      float nm = fmaxf(mr[r], mx);
      float al = __expf(mr[r] - nm);
      float p0 = __expf(s0[r] - nm);
      float p1 = __expf(s1[r] - nm);
      float rs = p0 + p1;
#pragma unroll
      for (int m = 1; m <= 8; m <<= 1) rs += __shfl_xor(rs, m, 32);
      lr[r] = lr[r]*al + rs;
      mr[r] = nm;
      o0[r] *= al; o1[r] *= al;
      Pbuf[wv][(r + 8*hi)*32 + lo]      = (_Float16)p0;   // re-layout C->A
      Pbuf[wv][(r + 8*hi)*32 + 16 + lo] = (_Float16)p1;
    }
    v16h ap = load_a16(&Pbuf[wv][lo*32], hi);   // same-wave LDS, dscnt-ordered
    // V fragments: vt is c-major so K pairs are contiguous -> dword loads
    v16h bv0 = load_b_ct(vtb + (size_t)(lo)*NSEQ + kb, hi);
    v16h bv1 = load_b_ct(vtb + (size_t)(16 + lo)*NSEQ + kb, hi);
    o0 = wmma_f16(ap, bv0, o0);
    o1 = wmma_f16(ap, bv1, o1);
  }
  // finalize: 1/l, gate, store f16
#pragma unroll
  for (int r = 0; r < 8; ++r) {
    float inv = 1.0f / lr[r];
    int qr = q0 + r + 8*hi;
    float g0 = (float)qb[(size_t)qr*QKG + 256 + h*CDIM + lo];
    float g1 = (float)qb[(size_t)qr*QKG + 256 + h*CDIM + 16 + lo];
    og[(size_t)(i*NSEQ + qr)*DP + h*CDIM + lo]      = (_Float16)(o0[r]*inv*g0);
    og[(size_t)(i*NSEQ + qr)*DP + h*CDIM + 16 + lo] = (_Float16)(o1[r]*inv*g1);
  }
}

// ---------------------------------------------------------------------------
// Kernel 3: output projection (WMMA) + out_bias + residual, f32 out.
// ---------------------------------------------------------------------------
__global__ __launch_bounds__(256) void k_out(
    const _Float16* __restrict__ og, const _Float16* __restrict__ Wopk,
    const float* __restrict__ obias, const float* __restrict__ Zraw,
    float* __restrict__ out) {
  __shared__ __align__(16) _Float16 Os[32][DP];
  const int tid = threadIdx.x;
  const int pt0 = blockIdx.x * 32;
  {  // cooperative load: 32x128 halves = 512 uint4
    const uint4* src = (const uint4*)(og + (size_t)pt0 * DP);
    uint4* dst = (uint4*)&Os[0][0];
    dst[tid] = src[tid];
    dst[tid + 256] = src[tid + 256];
  }
  __syncthreads();
  const int wv = tid >> 5, lane = tid & 31, lo = lane & 15, hi = lane >> 4;
  const int mt = wv >> 2;
  const int nt0 = (wv & 3) * 2;
  v8f a0 = zero8(), a1 = zero8();
  const _Float16* arow = &Os[mt*16 + lo][0];
#pragma unroll
  for (int ks = 0; ks < 4; ++ks) {
    v16h a  = load_a16(arow + ks*32, hi);
    v16h b0 = load_b_pk(Wopk + (size_t)(ks*16)*256, 256, nt0*16 + lo, hi);
    v16h b1 = load_b_pk(Wopk + (size_t)(ks*16)*256, 256, nt0*16 + 16 + lo, hi);
    a0 = wmma_f16(a, b0, a0);
    a1 = wmma_f16(a, b1, a1);
  }
  int c0 = nt0*16 + lo, c1 = c0 + 16;
  float ob0 = obias[c0], ob1 = obias[c1];
#pragma unroll
  for (int r = 0; r < 8; ++r) {
    size_t pt = (size_t)(pt0 + mt*16 + r + 8*hi);
    out[pt*DP + c0] = a0[r] + ob0 + Zraw[pt*DP + c0];
    out[pt*DP + c1] = a1[r] + ob1 + Zraw[pt*DP + c1];
  }
}

// ---------------------------------------------------------------------------
extern "C" void kernel_launch(void* const* d_in, const int* in_sizes, int n_in,
                              void* d_out, int out_size, void* d_ws, size_t ws_size,
                              hipStream_t stream) {
  (void)in_sizes; (void)n_in; (void)out_size; (void)ws_size;
  const float* Zraw  = (const float*)d_in[0];
  const float* Zmask = (const float*)d_in[1];
  const float* lnw   = (const float*)d_in[2];
  const float* lnb   = (const float*)d_in[3];
  const float* wb    = (const float*)d_in[4];
  const float* wq    = (const float*)d_in[5];
  const float* wk    = (const float*)d_in[6];
  const float* wvv   = (const float*)d_in[7];
  const float* wg    = (const float*)d_in[8];
  const float* bg    = (const float*)d_in[9];
  const float* wo    = (const float*)d_in[10];
  const float* obias = (const float*)d_in[11];
  float* out = (float*)d_out;

  // workspace carve (all chunks 256B-aligned)
  char* ws = (char*)d_ws;
  _Float16* Wpk  = (_Float16*)ws; ws += (size_t)128*512*2;        // 128 KiB
  _Float16* Wopk = (_Float16*)ws; ws += (size_t)128*128*2;        // 32 KiB
  float*    trib = (float*)ws;    ws += (size_t)NHEAD*NPTS*4;     // 1 MiB
  _Float16* qkvg = (_Float16*)ws; ws += (size_t)NPTS*QKG*2;       // 48 MiB
  _Float16* vt   = (_Float16*)ws; ws += (size_t)NPTS*DP*2;        // 16 MiB
  _Float16* og   = (_Float16*)ws; ws += (size_t)NPTS*DP*2;        // 16 MiB

  k_prep<<<(128*512 + 128*128 + 255)/256, 256, 0, stream>>>(wq, wk, wvv, wg, wo, Wpk, Wopk);
  k_proj<<<NPTS/32, 256, 0, stream>>>(Zraw, lnw, lnb, wb, bg, Wpk, qkvg, vt, trib);
  dim3 gattn(NSEQ, NHEAD);
  k_attn<<<gattn, 512, 0, stream>>>(qkvg, vt, trib, Zmask, og);
  k_out<<<NPTS/32, 256, 0, stream>>>(og, Wopk, obias, Zraw, out);
}
